// GraphormerEncoderLayer_13855564497052
// MI455X (gfx1250) — matrix operand order repaired and checked by reading, not compiled, not run on hardware
//
#include <hip/hip_runtime.h>

typedef __bf16 bf16_t;
typedef __attribute__((ext_vector_type(16))) __bf16 v16bf;
typedef __attribute__((ext_vector_type(8)))  __bf16 v8bf;
typedef __attribute__((ext_vector_type(8)))  float  v8f;

static constexpr int kN  = 3072;
static constexpr int kD  = 512;
static constexpr int kH  = 8;
static constexpr int kDK = 512;
static constexpr int kED = 64;
static constexpr int kL  = 5;
static constexpr int kP  = 196608;
static constexpr int kF  = 2048;
static constexpr float kNEG = -1000000.0f;
static constexpr float kSCALE = 0.04419417382415922f; // 1/sqrt(512)

union Frag { v16bf v; v8bf h[2]; };
union Acc  { v8f v; float f[8]; };

#define LDSTR 40   // bf16 elements per LDS row: 32 data + 8 pad (80B stride)

struct GemmArgs {
  const bf16_t* A; const bf16_t* B;
  int M, No, K, lda, ldb, ldo, ocol;
  float* fOut; bf16_t* bOut;
  const float* bias; const float* add1;
  const float* Bmat; float* Smat; const int* gid;
  float scale;
};

// ---------------------------------------------------------------------------
// NT GEMM: C[M,No] = A[M,K] * B[No,K]^T, bf16 inputs, f32 accumulate via WMMA.
// Block = 256 threads = 8 waves; tile 128x128; K step 32.
// EPI: 0 bf16 row-major (+bias)      (q,k projections)
//      1 bf16 transposed (+bias)     (v projection -> vT)
//      2 score epilogue (mask/NEG)   (q.k^T)
//      3 bf16 row-major, col offset  (p.v -> concat heads)
//      4 f32 (+bias +add1)           (out-proj + x ; ffn2 + x')
//      5 bf16 gelu(+bias)            (ffn1)
// ---------------------------------------------------------------------------
template<int EPI>
__global__ __launch_bounds__(256)
void gemm_nt(GemmArgs g) {
  __shared__ bf16_t lA[128 * LDSTR];
  __shared__ bf16_t lB[128 * LDSTR];
  const int tid  = threadIdx.x;
  const int wave = tid >> 5;
  const int lane = tid & 31;
  const int half = lane >> 4;
  const int rr   = lane & 15;
  const int m0 = blockIdx.y * 128;
  const int n0 = blockIdx.x * 128;

  Acc acc[8];
#pragma unroll
  for (int c = 0; c < 8; ++c)
#pragma unroll
    for (int i = 0; i < 8; ++i) acc[c].f[i] = 0.0f;

  const int ksteps = g.K / 32;
  for (int ks = 0; ks < ksteps; ++ks) {
    const int k0 = ks * 32;
    // stage A and B 128x32 bf16 tiles into LDS (each thread: 2x16B per tile)
#pragma unroll
    for (int i = 0; i < 2; ++i) {
      int idx = tid + i * 256;            // 0..511
      int row = idx >> 2, ch = idx & 3;   // 4 x 16B chunks per 64B row
      *(v8bf*)&lA[row * LDSTR + ch * 8] =
          *(const v8bf*)&g.A[(long)(m0 + row) * g.lda + k0 + ch * 8];
      *(v8bf*)&lB[row * LDSTR + ch * 8] =
          *(const v8bf*)&g.B[(long)(n0 + row) * g.ldb + k0 + ch * 8];
      if (ks + 1 < ksteps) {
        __builtin_prefetch(&g.A[(long)(m0 + row) * g.lda + k0 + 32 + ch * 8], 0, 1);
        __builtin_prefetch(&g.B[(long)(n0 + row) * g.ldb + k0 + 32 + ch * 8], 0, 1);
      }
    }
    __syncthreads();

    // A fragment: lane holds row (wave*16+rr); K chunks {0-7,16-23} / {8-15,24-31}
    Frag af;
    {
      const bf16_t* ap = &lA[(wave * 16 + rr) * LDSTR];
      af.h[0] = *(const v8bf*)(ap + half * 8);
      af.h[1] = *(const v8bf*)(ap + half * 8 + 16);
    }
#pragma unroll
    for (int c = 0; c < 8; ++c) {
      // B fragment: lane holds column (c*16+rr); K chunks {0-7,16-23} / {8-15,24-31}
      Frag bfg;
      const bf16_t* bp = &lB[(c * 16 + rr) * LDSTR];
      bfg.h[0] = *(const v8bf*)(bp + half * 8);
      bfg.h[1] = *(const v8bf*)(bp + half * 8 + 16);
      acc[c].v = __builtin_amdgcn_wmma_f32_16x16x32_bf16(
          false, af.v, false, bfg.v, (short)0, acc[c].v, false, false);
    }
    __syncthreads();
  }

  // ---- epilogue ----
  int gm[8];
  if (EPI == 2) {
#pragma unroll
    for (int gg = 0; gg < 8; ++gg)
      gm[gg] = g.gid[m0 + wave * 16 + half * 8 + gg];
  }
#pragma unroll
  for (int c = 0; c < 8; ++c) {
    const int n = n0 + c * 16 + rr;
    int gn = 0;
    if (EPI == 2) gn = g.gid[n];
    if (EPI == 1) {
      // transposed packed store: out[n*ldo + m..m+7]
      v8bf pk;
      float bb = g.bias[n];
#pragma unroll
      for (int gg = 0; gg < 8; ++gg) pk[gg] = (bf16_t)(acc[c].f[gg] + bb);
      int mbase = m0 + wave * 16 + half * 8;
      *(v8bf*)&g.bOut[(long)n * g.ldo + mbase] = pk;
    } else {
#pragma unroll
      for (int gg = 0; gg < 8; ++gg) {
        const int m = m0 + wave * 16 + half * 8 + gg;
        const float v = acc[c].f[gg];
        if (EPI == 0) {
          g.bOut[(long)m * g.ldo + n] = (bf16_t)(v + g.bias[n]);
        } else if (EPI == 2) {
          const long off = (long)m * g.ldo + n;
          const float a  = v * g.scale;
          const float bc = g.Bmat[off] + g.Smat[off];
          g.Smat[off] = (gm[gg] == gn) ? (a + bc) : (bc * kNEG);
        } else if (EPI == 3) {
          g.bOut[(long)m * g.ldo + n + g.ocol] = (bf16_t)v;
        } else if (EPI == 4) {
          const long off = (long)m * g.ldo + n;
          g.fOut[off] = v + g.bias[n] + g.add1[off];
        } else if (EPI == 5) {
          const float t = v + g.bias[n];
          g.bOut[(long)m * g.ldo + n] =
              (bf16_t)(0.5f * t * (1.0f + erff(t * 0.70710678118654752f)));
        }
      }
    }
  }
}

// ---------------------------------------------------------------------------
// transpose + f32->bf16 convert: out[z][c][r] = in[z][r][c]
// ---------------------------------------------------------------------------
__global__ __launch_bounds__(256)
void transpose_cvt(const float* in, bf16_t* out, int R, int C) {
  __shared__ float tile[32][33];
  const long base = (long)blockIdx.z * R * C;
  const int r0 = blockIdx.y * 32, c0 = blockIdx.x * 32;
  const int tx = threadIdx.x & 31, ty = threadIdx.x >> 5;
#pragma unroll
  for (int i = 0; i < 32; i += 8)
    tile[ty + i][tx] = in[base + (long)(r0 + ty + i) * C + (c0 + tx)];
  __syncthreads();
#pragma unroll
  for (int i = 0; i < 32; i += 8)
    out[base + (long)(c0 + ty + i) * R + (r0 + tx)] = (bf16_t)tile[tx][ty + i];
}

// ---------------------------------------------------------------------------
// layernorm over last dim + bf16 convert; one block per row
// ---------------------------------------------------------------------------
__global__ __launch_bounds__(256)
void layernorm_cvt(const float* x, const float* gw, const float* bw,
                   bf16_t* out, int Dn) {
  __shared__ float red[256];
  const long row = blockIdx.x;
  const float* xr = x + row * Dn;
  float s = 0.0f;
  for (int j = threadIdx.x; j < Dn; j += 256) s += xr[j];
  red[threadIdx.x] = s; __syncthreads();
  for (int st = 128; st; st >>= 1) {
    if (threadIdx.x < st) red[threadIdx.x] += red[threadIdx.x + st];
    __syncthreads();
  }
  const float mu = red[0] / Dn; __syncthreads();
  float v = 0.0f;
  for (int j = threadIdx.x; j < Dn; j += 256) { float d = xr[j] - mu; v += d * d; }
  red[threadIdx.x] = v; __syncthreads();
  for (int st = 128; st; st >>= 1) {
    if (threadIdx.x < st) red[threadIdx.x] += red[threadIdx.x + st];
    __syncthreads();
  }
  const float inv = rsqrtf(red[0] / Dn + 1e-5f);
  bf16_t* orow = out + row * Dn;
  for (int j = threadIdx.x; j < Dn; j += 256)
    orow[j] = (bf16_t)((xr[j] - mu) * inv * gw[j] + bw[j]);
}

// ---------------------------------------------------------------------------
// edge encoding: val[h][p] = mean over valid path steps of <edge_attr, ev[h][l]>
// one wave per path p
// ---------------------------------------------------------------------------
__global__ __launch_bounds__(256)
void edge_val_kernel(const float* edge_attr, const int* path_edges,
                     const int* path_len, const float* edge_vector,
                     float* val, int Pn) {
  __shared__ float ev[kH * kL * kED];
  for (int i = threadIdx.x; i < kH * kL * kED; i += 256) ev[i] = edge_vector[i];
  __syncthreads();
  const int wave = threadIdx.x >> 5, lane = threadIdx.x & 31;
  const int p = blockIdx.x * 8 + wave;
  if (p >= Pn) return;
  const int len = path_len[p];
  float acc[kH];
#pragma unroll
  for (int h = 0; h < kH; ++h) acc[h] = 0.0f;
  for (int l = 0; l < kL; ++l) {
    if (l < len) {
      const int e = path_edges[p * kL + l];
      const float a0 = edge_attr[(long)e * kED + lane];
      const float a1 = edge_attr[(long)e * kED + lane + 32];
#pragma unroll
      for (int h = 0; h < kH; ++h) {
        const float* evl = &ev[(h * kL + l) * kED];
        acc[h] += a0 * evl[lane] + a1 * evl[lane + 32];
      }
    }
  }
#pragma unroll
  for (int h = 0; h < kH; ++h)
    for (int o = 16; o; o >>= 1) acc[h] += __shfl_down(acc[h], o, 32);
  if (lane == 0) {
#pragma unroll
    for (int h = 0; h < kH; ++h)
      val[(long)h * Pn + p] = (len > 0) ? (acc[h] / (float)len) : 0.0f;
  }
}

__global__ void zero_kernel(float* p, long n) {
  long i = (long)blockIdx.x * 256 + threadIdx.x;
  if (i < n) p[i] = 0.0f;
}

__global__ void scatter_kernel(float* S, const float* val, const int* src,
                               const int* dst, int Pn, int Nn) {
  int p = blockIdx.x * 256 + threadIdx.x;
  if (p < Pn) S[(long)src[p] * Nn + dst[p]] = val[p];
}

// full-row softmax (denominator over ALL columns), then mask by same-group
__global__ __launch_bounds__(256)
void softmax_kernel(const float* S, bf16_t* Pr, const int* gid, int Nn) {
  __shared__ float red[256];
  const long row = blockIdx.x;
  const int gi = gid[row];
  const float* sr = S + row * Nn;
  float mx = -3.4e38f;
  for (int j = threadIdx.x; j < Nn; j += 256) mx = fmaxf(mx, sr[j]);
  red[threadIdx.x] = mx; __syncthreads();
  for (int st = 128; st; st >>= 1) {
    if (threadIdx.x < st) red[threadIdx.x] = fmaxf(red[threadIdx.x], red[threadIdx.x + st]);
    __syncthreads();
  }
  mx = red[0]; __syncthreads();
  float sum = 0.0f;
  for (int j = threadIdx.x; j < Nn; j += 256) sum += expf(sr[j] - mx);
  red[threadIdx.x] = sum; __syncthreads();
  for (int st = 128; st; st >>= 1) {
    if (threadIdx.x < st) red[threadIdx.x] += red[threadIdx.x + st];
    __syncthreads();
  }
  const float inv = 1.0f / red[0];
  bf16_t* pr = Pr + row * Nn;
  for (int j = threadIdx.x; j < Nn; j += 256)
    pr[j] = (bf16_t)((gid[j] == gi) ? expf(sr[j] - mx) * inv : 0.0f);
}

// ---------------------------------------------------------------------------
extern "C" void kernel_launch(void* const* d_in, const int* in_sizes, int n_in,
                              void* d_out, int out_size, void* d_ws, size_t ws_size,
                              hipStream_t stream) {
  const float* x          = (const float*)d_in[0];
  const float* edge_attr  = (const float*)d_in[1];
  const float* bmat       = (const float*)d_in[2];
  const int*   path_edges = (const int*)d_in[3];
  const int*   path_len   = (const int*)d_in[4];
  const int*   path_src   = (const int*)d_in[5];
  const int*   path_dst   = (const int*)d_in[6];
  const int*   gid        = (const int*)d_in[7];
  const float* Wq = (const float*)d_in[8];
  const float* bq = (const float*)d_in[9];
  const float* Wk = (const float*)d_in[10];
  const float* bk = (const float*)d_in[11];
  const float* Wv = (const float*)d_in[12];
  const float* bv = (const float*)d_in[13];
  const float* edge_vector = (const float*)d_in[14];
  const float* Wo = (const float*)d_in[15];
  const float* bo = (const float*)d_in[16];
  const float* ln1_g = (const float*)d_in[17];
  const float* ln1_b = (const float*)d_in[18];
  const float* ln2_g = (const float*)d_in[19];
  const float* ln2_b = (const float*)d_in[20];
  const float* W1 = (const float*)d_in[21];
  const float* b1 = (const float*)d_in[22];
  const float* W2 = (const float*)d_in[23];
  const float* b2 = (const float*)d_in[24];
  (void)in_sizes; (void)n_in; (void)out_size; (void)ws_size;

  char* base = (char*)d_ws; size_t off = 0;
  auto carve = [&](size_t bytes) -> char* {
    char* p = base + off; off += (bytes + 255) & ~(size_t)255; return p;
  };
  bf16_t* hbf = (bf16_t*)carve((size_t)kN * kD * 2);
  bf16_t* WqT = (bf16_t*)carve((size_t)kH * kD * kDK * 2);
  bf16_t* WkT = (bf16_t*)carve((size_t)kH * kD * kDK * 2);
  bf16_t* WvT = (bf16_t*)carve((size_t)kH * kD * kDK * 2);
  bf16_t* WoT = (bf16_t*)carve((size_t)kH * kDK * kD * 2);
  bf16_t* W1T = (bf16_t*)carve((size_t)kD * kF * 2);
  bf16_t* W2T = (bf16_t*)carve((size_t)kF * kD * 2);
  bf16_t* qb  = (bf16_t*)carve((size_t)kH * kN * kDK * 2);
  bf16_t* kb  = (bf16_t*)carve((size_t)kH * kN * kDK * 2);
  bf16_t* vT  = (bf16_t*)carve((size_t)kH * kDK * kN * 2);
  float*  val = (float*) carve((size_t)kH * kP * 4);
  float*  S   = (float*) carve((size_t)kN * kN * 4);
  bf16_t* Ppb = (bf16_t*)carve((size_t)kN * kN * 2);
  bf16_t* ohc = (bf16_t*)carve((size_t)kN * kH * kDK * 2);
  float*  xpr = (float*) carve((size_t)kN * kD * 4);
  bf16_t* h2b = (bf16_t*)carve((size_t)kN * kD * 2);
  bf16_t* tb  = (bf16_t*)carve((size_t)kN * kF * 2);

  const dim3 blk(256);

  // weights -> transposed bf16
  transpose_cvt<<<dim3(kDK / 32, kD / 32, kH), blk, 0, stream>>>(Wq, WqT, kD, kDK);
  transpose_cvt<<<dim3(kDK / 32, kD / 32, kH), blk, 0, stream>>>(Wk, WkT, kD, kDK);
  transpose_cvt<<<dim3(kDK / 32, kD / 32, kH), blk, 0, stream>>>(Wv, WvT, kD, kDK);
  transpose_cvt<<<dim3(kD / 32, (kH * kDK) / 32, 1), blk, 0, stream>>>(Wo, WoT, kH * kDK, kD);
  transpose_cvt<<<dim3(kF / 32, kD / 32, 1), blk, 0, stream>>>(W1, W1T, kD, kF);
  transpose_cvt<<<dim3(kD / 32, kF / 32, 1), blk, 0, stream>>>(W2, W2T, kF, kD);

  // LN1
  layernorm_cvt<<<kN, blk, 0, stream>>>(x, ln1_g, ln1_b, hbf, kD);

  // QKV projections (24 GEMMs)
  for (int h = 0; h < kH; ++h) {
    GemmArgs gq = {};
    gq.A = hbf; gq.lda = kD; gq.M = kN; gq.No = kDK; gq.K = kD;
    gq.B = WqT + (size_t)h * kDK * kD; gq.ldb = kD;
    gq.bOut = qb + (size_t)h * kN * kDK; gq.ldo = kDK; gq.bias = bq + h * kDK;
    gemm_nt<0><<<dim3(kDK / 128, kN / 128), blk, 0, stream>>>(gq);

    GemmArgs gk = gq;
    gk.B = WkT + (size_t)h * kDK * kD;
    gk.bOut = kb + (size_t)h * kN * kDK; gk.bias = bk + h * kDK;
    gemm_nt<0><<<dim3(kDK / 128, kN / 128), blk, 0, stream>>>(gk);

    GemmArgs gv = gq;
    gv.B = WvT + (size_t)h * kDK * kD;
    gv.bOut = vT + (size_t)h * kDK * kN; gv.ldo = kN; gv.bias = bv + h * kDK;
    gemm_nt<1><<<dim3(kDK / 128, kN / 128), blk, 0, stream>>>(gv);
  }

  // edge encoding values val[H][P]
  edge_val_kernel<<<kP / 8, blk, 0, stream>>>(edge_attr, path_edges, path_len,
                                              edge_vector, val, kP);

  // attention, one head at a time (reuse S / Ppb buffers)
  const long nn = (long)kN * kN;
  for (int h = 0; h < kH; ++h) {
    zero_kernel<<<(unsigned)((nn + 255) / 256), blk, 0, stream>>>(S, nn);
    scatter_kernel<<<kP / 256, blk, 0, stream>>>(S, val + (size_t)h * kP,
                                                 path_src, path_dst, kP, kN);
    GemmArgs gs = {};
    gs.A = qb + (size_t)h * kN * kDK; gs.B = kb + (size_t)h * kN * kDK;
    gs.M = kN; gs.No = kN; gs.K = kDK; gs.lda = kDK; gs.ldb = kDK; gs.ldo = kN;
    gs.Bmat = bmat; gs.Smat = S; gs.gid = gid; gs.scale = kSCALE;
    gemm_nt<2><<<dim3(kN / 128, kN / 128), blk, 0, stream>>>(gs);

    softmax_kernel<<<kN, blk, 0, stream>>>(S, Ppb, gid, kN);

    GemmArgs gp = {};
    gp.A = Ppb; gp.lda = kN; gp.B = vT + (size_t)h * kDK * kN; gp.ldb = kN;
    gp.M = kN; gp.No = kDK; gp.K = kN;
    gp.bOut = ohc; gp.ldo = kH * kDK; gp.ocol = h * kDK;
    gemm_nt<3><<<dim3(kDK / 128, kN / 128), blk, 0, stream>>>(gp);
  }

  // output projection + residual -> x'
  {
    GemmArgs go = {};
    go.A = ohc; go.lda = kH * kDK; go.B = WoT; go.ldb = kH * kDK;
    go.M = kN; go.No = kD; go.K = kH * kDK;
    go.fOut = xpr; go.ldo = kD; go.bias = bo; go.add1 = x;
    gemm_nt<4><<<dim3(kD / 128, kN / 128), blk, 0, stream>>>(go);
  }
  // LN2
  layernorm_cvt<<<kN, blk, 0, stream>>>(xpr, ln2_g, ln2_b, h2b, kD);
  // FFN1 (gelu)
  {
    GemmArgs g1 = {};
    g1.A = h2b; g1.lda = kD; g1.B = W1T; g1.ldb = kD;
    g1.M = kN; g1.No = kF; g1.K = kD;
    g1.bOut = tb; g1.ldo = kF; g1.bias = b1;
    gemm_nt<5><<<dim3(kF / 128, kN / 128), blk, 0, stream>>>(g1);
  }
  // FFN2 + residual -> d_out
  {
    GemmArgs g2 = {};
    g2.A = tb; g2.lda = kF; g2.B = W2T; g2.ldb = kF;
    g2.M = kN; g2.No = kD; g2.K = kF;
    g2.fOut = (float*)d_out; g2.ldo = kD; g2.bias = b2; g2.add1 = xpr;
    gemm_nt<4><<<dim3(kD / 128, kN / 128), blk, 0, stream>>>(g2);
  }
}
